// TritonScatterConv_25451976196327
// MI455X (gfx1250) — matrix operand order repaired and checked by reading, not compiled, not run on hardware
//
#include <hip/hip_runtime.h>
#include <hip/hip_bf16.h>
#include <math.h>

// ---------------------------------------------------------------------------
// Problem constants (from reference)
// ---------------------------------------------------------------------------
#define BB   2
#define LL   2048
#define CC   1024
#define HH   8
#define POSN 16
#define DD   (CC / HH)        // 128
#define SS   33               // MAXS+1 sample positions, stride -16..16
#define BL   (BB * LL)        // 4096 rows
#define MAXF 16.0f
#define MINF 1.0f
#define SCALE 0.25f           // POS^-0.5

typedef __bf16  bf16;
typedef __bf16  bf16x8  __attribute__((ext_vector_type(8)));
typedef __bf16  bf16x16 __attribute__((ext_vector_type(16)));
typedef float   floatx8 __attribute__((ext_vector_type(8)));

__device__ __forceinline__ float sigmoidf_(float v) { return 1.0f / (1.0f + __expf(-v)); }
__device__ __forceinline__ float siluf_(float v)    { return v * sigmoidf_(v); }

union Frag { bf16x16 v; struct { bf16x8 lo, hi; } p; };

// A 16x32 fragment (CDNA5 ISA 7.12.2): lane m = lane&15, kg = lane>>4;
// elements = K[k+8kg .. +8) ++ K[k+16+8kg .. +8)  -> two contiguous 16B loads
__device__ __forceinline__ bf16x16 load_afrag(const bf16* __restrict__ p, int k, int kg) {
    Frag f;
    f.p.lo = *reinterpret_cast<const bf16x8*>(p + k + 8 * kg);
    f.p.hi = *reinterpret_cast<const bf16x8*>(p + k + 16 + 8 * kg);
    return f.v;
}
// B 32x16 fragment: lane n = lane&15, kg = lane>>4; elements = K[k+8kg .. +8) ++ K[k+16+8kg .. +8)
__device__ __forceinline__ bf16x16 load_bfrag(const bf16* __restrict__ p, int k, int kg) {
    Frag f;
    f.p.lo = *reinterpret_cast<const bf16x8*>(p + k + 8 * kg);
    f.p.hi = *reinterpret_cast<const bf16x8*>(p + k + 16 + 8 * kg);
    return f.v;
}

// ---------------------------------------------------------------------------
// Elementwise f32 -> bf16 convert (activations)
// ---------------------------------------------------------------------------
__global__ void k_to_bf16(const float* __restrict__ in, bf16* __restrict__ out, int n) {
    for (int i = blockIdx.x * blockDim.x + threadIdx.x; i < n; i += gridDim.x * blockDim.x)
        out[i] = (bf16)in[i];
}

// ---------------------------------------------------------------------------
// Weight transpose + convert: in (K,N) f32 row-major -> out (N,K) bf16
// ---------------------------------------------------------------------------
__global__ void k_transpose_bf16(const float* __restrict__ in, bf16* __restrict__ out,
                                 int K, int N) {
    int total = K * N;
    int i = blockIdx.x * blockDim.x + threadIdx.x;
    if (i >= total) return;
    int n = i / K, k = i - n * K;
    out[i] = (bf16)in[(size_t)k * N + n];
}

// ---------------------------------------------------------------------------
// Wave-parameter kernel: one wave32 per row, 8 waves per block.
// ---------------------------------------------------------------------------
__global__ void k_waveparams(const float* __restrict__ x,
                             const float* __restrict__ w_wave,
                             const float* __restrict__ b_wave,
                             float* __restrict__ favg,
                             float* __restrict__ pavg,
                             float* __restrict__ davg) {
    int wave = threadIdx.x >> 5, lane = threadIdx.x & 31;
    int row  = blockIdx.x * 8 + wave;
    if (row >= BL) return;
    const float* xr = x + (size_t)row * CC;

    float mine = 0.0f;
    for (int j = 0; j < 24; ++j) {
        float p = 0.0f;
        for (int k = lane; k < CC; k += 32)
            p = fmaf(xr[k], w_wave[k * 24 + j], p);
        #pragma unroll
        for (int off = 16; off > 0; off >>= 1)
            p += __shfl_xor(p, off, 32);
        if (lane == j) mine = p;
    }
    if (lane < 24) {
        float v = mine + b_wave[lane];
        float w = siluf_(v);
        int   g = lane >> 3;
        float t;
        if (g == 0)      t = sigmoidf_(w) * (MAXF - MINF) + MINF;   // freq
        else if (g == 1) t = tanhf(w) * MAXF;                       // phase
        else             t = sigmoidf_(w) * 9.5f + 0.5f;            // decay
        t += __shfl_xor(t, 1, 32);
        t += __shfl_xor(t, 2, 32);
        t += __shfl_xor(t, 4, 32);
        t *= 0.125f;                                                // mean over H=8
        if ((lane & 7) == 0) {
            if (g == 0)      favg[row] = t;
            else if (g == 1) pavg[row] = t;
            else             davg[row] = t;
        }
    }
}

// ---------------------------------------------------------------------------
// WMMA bf16 GEMM: C[M,N] = epilogue(A[M,K] @ Bt[N,K]^T + bias).
// Register-blocked: each wave owns a 32x64 tile (2 M-frags x 4 N-frags,
// 8 f32 accumulators => 8 v_wmma per 32-deep k-step, fragment traffic
// 0.75KB/WMMA). __launch_bounds__(256,1) caps occupancy at 2 waves/SIMD so
// the ~130 live VGPRs fit WITHOUT scratch spills; latency is hidden by the
// 8 independent WMMAs + 12 outstanding b128 loads per iteration (unroll 2).
// 8 waves / 256-thread block. Requires M%32==0, N%64==0, K%32==0.
//
// EPI==0: silu(v+bias);  EPI==1: sigmoid(v+bias) * mulsrc[idx]
// ---------------------------------------------------------------------------
template <int EPI>
__global__ void __launch_bounds__(256, 1)
k_wmma_gemm(const bf16* __restrict__ A,
            const bf16* __restrict__ Bt,
            const float* __restrict__ bias,
            float* __restrict__ Cf,
            bf16*  __restrict__ Cb,
            const float* __restrict__ mulsrc,
            int M, int N, int K) {
    const int wave = threadIdx.x >> 5;
    const int lane = threadIdx.x & 31;
    const int tilesN = N >> 6;                 // 64-wide super-tiles
    const int tiles  = (M >> 5) * tilesN;      // 32-tall super-tiles
    const int tile   = blockIdx.x * 8 + wave;
    if (tile >= tiles) return;

    const int tm = tile / tilesN, tn = tile - tm * tilesN;
    const int row0 = tm << 5, col0 = tn << 6;
    const int m  = lane & 15;
    const int kg = lane >> 4;

    const bf16* Ap0 = A  + (size_t)(row0 + m) * K;
    const bf16* Ap1 = A  + (size_t)(row0 + 16 + m) * K;
    const bf16* Bp0 = Bt + (size_t)(col0 + m) * K;
    const bf16* Bp1 = Bt + (size_t)(col0 + 16 + m) * K;
    const bf16* Bp2 = Bt + (size_t)(col0 + 32 + m) * K;
    const bf16* Bp3 = Bt + (size_t)(col0 + 48 + m) * K;

    floatx8 acc[8];
    #pragma unroll
    for (int i = 0; i < 8; ++i) acc[i] = (floatx8){0.f,0.f,0.f,0.f,0.f,0.f,0.f,0.f};

    #pragma unroll 2
    for (int k = 0; k < K; k += 32) {
        // WGP-scope prefetch two steps ahead (reuses live base addresses)
        __builtin_prefetch(Ap0 + k + 64, 0, 3);
        __builtin_prefetch(Bp0 + k + 64, 0, 3);
        const bf16x16 a0 = load_afrag(Ap0, k, kg);
        const bf16x16 a1 = load_afrag(Ap1, k, kg);
        const bf16x16 b0 = load_bfrag(Bp0, k, kg);
        const bf16x16 b1 = load_bfrag(Bp1, k, kg);
        const bf16x16 b2 = load_bfrag(Bp2, k, kg);
        const bf16x16 b3 = load_bfrag(Bp3, k, kg);
        acc[0] = __builtin_amdgcn_wmma_f32_16x16x32_bf16(false, a0, false, b0, (short)0, acc[0], false, false);
        acc[1] = __builtin_amdgcn_wmma_f32_16x16x32_bf16(false, a0, false, b1, (short)0, acc[1], false, false);
        acc[2] = __builtin_amdgcn_wmma_f32_16x16x32_bf16(false, a0, false, b2, (short)0, acc[2], false, false);
        acc[3] = __builtin_amdgcn_wmma_f32_16x16x32_bf16(false, a0, false, b3, (short)0, acc[3], false, false);
        acc[4] = __builtin_amdgcn_wmma_f32_16x16x32_bf16(false, a1, false, b0, (short)0, acc[4], false, false);
        acc[5] = __builtin_amdgcn_wmma_f32_16x16x32_bf16(false, a1, false, b1, (short)0, acc[5], false, false);
        acc[6] = __builtin_amdgcn_wmma_f32_16x16x32_bf16(false, a1, false, b2, (short)0, acc[6], false, false);
        acc[7] = __builtin_amdgcn_wmma_f32_16x16x32_bf16(false, a1, false, b3, (short)0, acc[7], false, false);
    }

    // Epilogue: C/D layout => VGPR r -> row r + 8*kg, col = lane&15
    float bv[4];
    #pragma unroll
    for (int nt = 0; nt < 4; ++nt)
        bv[nt] = bias ? bias[col0 + nt * 16 + m] : 0.0f;

    #pragma unroll
    for (int mt = 0; mt < 2; ++mt) {
        #pragma unroll
        for (int nt = 0; nt < 4; ++nt) {
            const floatx8 a = acc[mt * 4 + nt];
            const int nout = col0 + nt * 16 + m;
            #pragma unroll
            for (int r = 0; r < 8; ++r) {
                int    mout = row0 + mt * 16 + r + 8 * kg;
                size_t idx  = (size_t)mout * N + nout;
                float  v    = a[r] + bv[nt];
                float  o;
                if (EPI == 0) o = siluf_(v);
                else          o = sigmoidf_(v) * mulsrc[idx];
                if (Cf) Cf[idx] = o;
                if (Cb) Cb[idx] = (bf16)o;
            }
        }
    }
}

// ---------------------------------------------------------------------------
// Attention kernel: one block (256 thr) per (b,l) row.
// ---------------------------------------------------------------------------
__global__ void k_attention(const float* __restrict__ x,
                            const float* __restrict__ q,      // (BL,128) silu'd
                            const float* __restrict__ kw,     // key_weight (16)
                            const float* __restrict__ favg,
                            const float* __restrict__ pavg,
                            const float* __restrict__ davg,
                            float* __restrict__ out0f,
                            bf16*  __restrict__ out0b) {
    const int bl = blockIdx.x;
    const int b  = bl / LL, l = bl - b * LL;
    const int tid = threadIdx.x;

    __shared__ float s_attn[HH][SS + 3];
    __shared__ float s_rel[SS];
    __shared__ int   s_idx[SS];
    __shared__ int   s_val[SS];
    __shared__ float s_qk[HH];

    const float fa = favg[bl], pa = pavg[bl], da = davg[bl];

    if (tid < SS) {
        float st = (float)(tid - 16);
        float sp = (float)l + st * fa + pa;
        int   v  = (sp >= 0.0f) && (sp < (float)LL);
        int   ix = (int)sp;
        ix = ix < 0 ? 0 : (ix > LL - 1 ? LL - 1 : ix);
        s_idx[tid] = ix;
        s_val[tid] = v;
        s_rel[tid] = fabsf(st) * fa;
    }
    if (tid < HH) {
        const float* qr = q + (size_t)bl * (HH * POSN) + tid * POSN;
        float s = 0.0f;
        #pragma unroll
        for (int p = 0; p < POSN; ++p) s = fmaf(qr[p], kw[p], s);
        s_qk[tid] = s;
    }
    __syncthreads();

    if (tid < HH) {
        const float qkh = s_qk[tid];
        const float dcl = fmaxf(da, 0.1f);
        float mx = -3.0e38f;
        for (int s = 0; s < SS; ++s)
            if (s_val[s]) mx = fmaxf(mx, qkh * s_rel[s] * SCALE);
        float sum = 0.0f;
        for (int s = 0; s < SS; ++s) {
            float e = s_val[s] ? __expf(qkh * s_rel[s] * SCALE - mx) : 0.0f;
            s_attn[tid][s] = e;
            sum += e;
        }
        float isum = sum > 0.0f ? 1.0f / sum : 0.0f;
        float tot = 0.0f;
        for (int s = 0; s < SS; ++s) {
            float a = s_attn[tid][s] * isum;
            a *= s_val[s] ? __expf(-s_rel[s] / dcl) : 0.0f;
            s_attn[tid][s] = a;
            tot += a;
        }
        float inv = 1.0f / (tot + 1e-8f);
        for (int s = 0; s < SS; ++s) s_attn[tid][s] *= inv;
    }
    __syncthreads();

    const float* xb = x + (size_t)b * LL * CC;
    for (int c = tid; c < CC; c += blockDim.x) {
        const int h = c >> 7;                 // c / D
        float acc = 0.0f;
        #pragma unroll 4
        for (int s = 0; s < SS; ++s)
            acc = fmaf(s_attn[h][s], xb[(size_t)s_idx[s] * CC + c], acc);
        size_t o = (size_t)bl * CC + c;
        out0f[o] = acc;
        out0b[o] = (bf16)acc;
    }
}

// ---------------------------------------------------------------------------
// Host launcher
// ---------------------------------------------------------------------------
extern "C" void kernel_launch(void* const* d_in, const int* in_sizes, int n_in,
                              void* d_out, int out_size, void* d_ws, size_t ws_size,
                              hipStream_t stream) {
    const float* x    = (const float*)d_in[0];
    const float* wwv  = (const float*)d_in[1];
    const float* bwv  = (const float*)d_in[2];
    const float* wq   = (const float*)d_in[3];
    const float* bq   = (const float*)d_in[4];
    const float* kw   = (const float*)d_in[5];
    const float* wse1 = (const float*)d_in[6];
    const float* bse1 = (const float*)d_in[7];
    const float* wse2 = (const float*)d_in[8];
    const float* bse2 = (const float*)d_in[9];
    const float* wout = (const float*)d_in[10];
    float* out = (float*)d_out;

    // -------- workspace carve-out (256B aligned) --------
    char*  base = (char*)d_ws;
    size_t off  = 0;
    auto alloc = [&](size_t bytes) -> void* {
        off = (off + 255) & ~(size_t)255;
        void* p = base + off;
        off += bytes;
        return p;
    };
    bf16*  xbf    = (bf16*) alloc((size_t)BL * CC * 2);
    bf16*  wq_t   = (bf16*) alloc((size_t)128 * CC * 2);       // (128,1024)
    bf16*  wse1_t = (bf16*) alloc((size_t)256 * CC * 2);       // (256,1024)
    bf16*  wse2_t = (bf16*) alloc((size_t)CC * 256 * 2);       // (1024,256)
    bf16*  wout_t = (bf16*) alloc((size_t)CC * CC * 2);        // (1024,1024)
    float* favg   = (float*)alloc((size_t)BL * 4);
    float* pavg   = (float*)alloc((size_t)BL * 4);
    float* davg   = (float*)alloc((size_t)BL * 4);
    float* qbuf   = (float*)alloc((size_t)BL * 128 * 4);
    float* out0f  = (float*)alloc((size_t)BL * CC * 4);
    bf16*  out0b  = (bf16*) alloc((size_t)BL * CC * 2);
    bf16*  h1b    = (bf16*) alloc((size_t)BL * 256 * 2);
    bf16*  out1b  = (bf16*) alloc((size_t)BL * CC * 2);
    (void)ws_size; (void)in_sizes; (void)n_in; (void)out_size;

    // 1) convert activations + (transposed) weights to bf16
    {
        int n = BL * CC;
        k_to_bf16<<<2048, 256, 0, stream>>>(x, xbf, n);
        k_transpose_bf16<<<(CC * 128  + 255) / 256, 256, 0, stream>>>(wq,   wq_t,   CC,  128);
        k_transpose_bf16<<<(CC * 256  + 255) / 256, 256, 0, stream>>>(wse1, wse1_t, CC,  256);
        k_transpose_bf16<<<(256 * CC  + 255) / 256, 256, 0, stream>>>(wse2, wse2_t, 256, CC);
        k_transpose_bf16<<<(CC * CC   + 255) / 256, 256, 0, stream>>>(wout, wout_t, CC,  CC);
    }

    // 2) per-row wave parameters (freq/phase/decay head-averages)
    k_waveparams<<<BL / 8, 256, 0, stream>>>(x, wwv, bwv, favg, pavg, davg);

    // 3) q = silu(x @ w_query + b_query)      [4096x1024 @ 1024x128]
    {
        int tiles = (BL / 32) * (128 / 64);
        k_wmma_gemm<0><<<(tiles + 7) / 8, 256, 0, stream>>>(
            xbf, wq_t, bq, qbuf, (bf16*)nullptr, (const float*)nullptr, BL, 128, CC);
    }

    // 4) strided-gather attention -> out0
    k_attention<<<BL, 256, 0, stream>>>(x, qbuf, kw, favg, pavg, davg, out0f, out0b);

    // 5) h1 = silu(out0 @ w_se1 + b_se1)      [4096x1024 @ 1024x256]
    {
        int tiles = (BL / 32) * (256 / 64);
        k_wmma_gemm<0><<<(tiles + 7) / 8, 256, 0, stream>>>(
            out0b, wse1_t, bse1, (float*)nullptr, h1b, (const float*)nullptr, BL, 256, CC);
    }

    // 6) out1 = out0 * sigmoid(h1 @ w_se2 + b_se2)   [4096x256 @ 256x1024]
    {
        int tiles = (BL / 32) * (CC / 64);
        k_wmma_gemm<1><<<(tiles + 7) / 8, 256, 0, stream>>>(
            h1b, wse2_t, bse2, (float*)nullptr, out1b, out0f, BL, CC, 256);
    }

    // 7) out = silu(out1 @ w_out)             [4096x1024 @ 1024x1024]
    {
        int tiles = (BL / 32) * (CC / 64);
        k_wmma_gemm<0><<<(tiles + 7) / 8, 256, 0, stream>>>(
            out1b, wout_t, (const float*)nullptr, out, (bf16*)nullptr,
            (const float*)nullptr, BL, CC, CC);
    }
}
